// VQVAE_40931038331442
// MI455X (gfx1250) — matrix operand order repaired and checked by reading, not compiled, not run on hardware
//
#include <hip/hip_runtime.h>
#include <cstdint>
#include <cstddef>

// ---------------------------------------------------------------------------
// VQ-VAE forward for MI455X (gfx1250).
// Big GEMMs: LDS-blocked double-buffered f16 WMMA (128x64 tile, 4 wmma/wave/step),
// staging issues all global_load_b128s before waiting (MLP-friendly).
// Attention GEMMs: direct batched WMMA. Wave32 row reductions for LN/softmax/VQ.
// ---------------------------------------------------------------------------

typedef __attribute__((ext_vector_type(16))) _Float16 v16h;
typedef __attribute__((ext_vector_type(8)))  _Float16 v8h;
typedef __attribute__((ext_vector_type(4)))  _Float16 v4h;
typedef __attribute__((ext_vector_type(8)))  float    v8f;

constexpr int B_ = 16, S_ = 256, D_ = 256, H_ = 8, HD_ = 32;
constexpr int K_ = 8192, DEPTH_ = 4, DFF_ = 1024;
constexpr int MROWS = B_ * S_;            // 4096 token rows
typedef long long ll;

__device__ __forceinline__ float gelu_tanh(float x)
{
    return 0.5f * x * (1.0f + tanhf(0.7978845608f * (x + 0.044715f * x * x * x)));
}

// ---------------------------------------------------------------------------
// LDS-blocked GEMM: C = epi(alpha * A x B (+ Res)).
// Requirements: M % 128 == 0, N % 64 == 0, Kd % 32 == 0, lda/ldb % 4 == 0.
// TRANSB: B stored [N, Kd] (row = output column), else [Kd, N].
// LDS layout matches CDNA5 WMMA fragment layouts so fragment reads are
// contiguous 16B ds_load_b128:
//   As[row][k]   (f16)  : A-frag lane reads k[kh*8..+7] and k[kh*8+16..+7]
//   Bs[col][k]   (f16)  : B-frag lane reads k[kh*8..+7] and k[kh*8+16..+7]
// ---------------------------------------------------------------------------
template<int EPI, bool TRANSB>
__global__ __launch_bounds__(256)
void gemm_wmma_lds(const float* __restrict__ A, const float* __restrict__ Bm,
                   const float* __restrict__ Res, float* __restrict__ C,
                   int M, int N, int Kd, int lda, int ldb, int ldc, float alpha)
{
    __shared__ __align__(16) _Float16 As[2][128][32];   // 16 KB
    __shared__ __align__(16) _Float16 Bs[2][64][32];    //  8 KB (transposed: [n][k])

    const int t     = threadIdx.x;
    const int lane  = t & 31;
    const int wave  = t >> 5;
    const int wm    = wave >> 1;        // 0..3 -> 32-row slice
    const int wn    = wave & 1;         // 0..1 -> 32-col slice
    const int m0    = blockIdx.y * 128;
    const int n0    = blockIdx.x * 64;
    const int khalf = lane >> 4;
    const int l16   = lane & 15;

    // Staging: issue ALL global_load_b128s first (keeps 6 loads in flight per
    // thread; compiler drains with decreasing s_wait_loadcnt), then cvt+ds_store.
    auto stage = [&](int k0, int pp) {
        float4 fa[4], fb[2];
#pragma unroll
        for (int i = 0; i < 4; ++i) {
            const int c  = t + i * 256;          // 1024 chunks of 4 floats
            const int r  = c >> 3;
            const int c4 = (c & 7) * 4;
            fa[i] = *(const float4*)&A[(size_t)(m0 + r) * lda + k0 + c4];
        }
#pragma unroll
        for (int i = 0; i < 2; ++i) {
            const int c = t + i * 256;           // 512 chunks
            if constexpr (TRANSB) {
                const int n  = c >> 3;
                const int k4 = (c & 7) * 4;
                fb[i] = *(const float4*)&Bm[(size_t)(n0 + n) * ldb + k0 + k4];
            } else {
                const int k  = c >> 4;
                const int n4 = (c & 15) * 4;
                fb[i] = *(const float4*)&Bm[(size_t)(k0 + k) * ldb + n0 + n4];
            }
        }
#pragma unroll
        for (int i = 0; i < 4; ++i) {
            const int c  = t + i * 256;
            const int r  = c >> 3;
            const int c4 = (c & 7) * 4;
            v4h h;
            h[0] = (_Float16)fa[i].x; h[1] = (_Float16)fa[i].y;
            h[2] = (_Float16)fa[i].z; h[3] = (_Float16)fa[i].w;
            *(v4h*)&As[pp][r][c4] = h;
        }
#pragma unroll
        for (int i = 0; i < 2; ++i) {
            const int c = t + i * 256;
            if constexpr (TRANSB) {
                const int n  = c >> 3;
                const int k4 = (c & 7) * 4;
                v4h h;
                h[0] = (_Float16)fb[i].x; h[1] = (_Float16)fb[i].y;
                h[2] = (_Float16)fb[i].z; h[3] = (_Float16)fb[i].w;
                *(v4h*)&Bs[pp][n][k4] = h;
            } else {
                const int k  = c >> 4;
                const int n4 = (c & 15) * 4;
                Bs[pp][n4 + 0][k] = (_Float16)fb[i].x;
                Bs[pp][n4 + 1][k] = (_Float16)fb[i].y;
                Bs[pp][n4 + 2][k] = (_Float16)fb[i].z;
                Bs[pp][n4 + 3][k] = (_Float16)fb[i].w;
            }
        }
    };

    v8f acc[2][2] = {};
    const int nK = Kd / 32;

    stage(0, 0);
    __syncthreads();

    for (int kt = 0; kt < nK; ++kt) {
        const int pp = kt & 1;
        if (kt + 1 < nK) stage((kt + 1) * 32, pp ^ 1);

        v16h afrag[2], bfrag[2];
#pragma unroll
        for (int i = 0; i < 2; ++i) {
            const int row = wm * 32 + i * 16 + l16;
            const v8h lo = *(const v8h*)&As[pp][row][khalf * 8];
            const v8h hi = *(const v8h*)&As[pp][row][khalf * 8 + 16];
#pragma unroll
            for (int e = 0; e < 8; ++e) { afrag[i][e] = lo[e]; afrag[i][e + 8] = hi[e]; }
        }
#pragma unroll
        for (int j = 0; j < 2; ++j) {
            const int col = wn * 32 + j * 16 + l16;
            const v8h lo = *(const v8h*)&Bs[pp][col][khalf * 8];
            const v8h hi = *(const v8h*)&Bs[pp][col][khalf * 8 + 16];
#pragma unroll
            for (int e = 0; e < 8; ++e) { bfrag[j][e] = lo[e]; bfrag[j][e + 8] = hi[e]; }
        }
#pragma unroll
        for (int i = 0; i < 2; ++i)
#pragma unroll
            for (int j = 0; j < 2; ++j)
                acc[i][j] = __builtin_amdgcn_wmma_f32_16x16x32_f16(
                    false, afrag[i], false, bfrag[j], (short)0, acc[i][j], false, false);

        __syncthreads();
    }

    // Epilogue: C layout -> VGPR r = row (sub)m0 + r + 8*khalf, col = l16.
#pragma unroll
    for (int i = 0; i < 2; ++i)
#pragma unroll
        for (int j = 0; j < 2; ++j)
#pragma unroll
            for (int r = 0; r < 8; ++r) {
                const int m = m0 + wm * 32 + i * 16 + r + 8 * khalf;
                const int n = n0 + wn * 32 + j * 16 + l16;
                float val = alpha * acc[i][j][r];
                if constexpr (EPI == 1) val += Res[(size_t)m * ldc + n];
                if constexpr (EPI == 2) val = gelu_tanh(val);
                C[(size_t)m * ldc + n] = val;
            }
}

// ---------------------------------------------------------------------------
// Direct batched GEMM for attention (small K / N per head). One 16x16 tile/wave.
// ---------------------------------------------------------------------------
template<int EPI, bool TRANSB>
__global__ __launch_bounds__(256)
void gemm_wmma(const float* __restrict__ A, const float* __restrict__ Bm,
               const float* __restrict__ Res, float* __restrict__ C,
               int M, int N, int Kd, int lda, int ldb, int ldc, int nh,
               ll sAb, ll sAh, ll sBb, ll sBh, ll sCb, ll sCh, ll sRb, ll sRh,
               float alpha)
{
    const int lane = threadIdx.x & 31;
    const int wave = threadIdx.x >> 5;
    const int m0 = blockIdx.y * 128 + wave * 16;
    const int n0 = blockIdx.x * 16;
    if (m0 >= M || n0 >= N) return;           // wave-uniform guard (EXEC stays full)

    const int z  = blockIdx.z;
    const int bb = z / nh, hh = z % nh;
    const float* Ap = A  + bb * sAb + hh * sAh;
    const float* Bp = Bm + bb * sBb + hh * sBh;
    float*       Cp = C  + bb * sCb + hh * sCh;

    const int mA = m0 + (lane & 15);
    const int kA = (lane >> 4) * 8;
    const int nB = n0 + (lane & 15);
    const int kB = (lane >> 4) * 8;

    v8f acc = {};
    for (int k0 = 0; k0 < Kd; k0 += 32) {
        if (k0 + 32 < Kd)   // gfx1250 global_prefetch_b8 on the next A panel
            __builtin_prefetch(&Ap[(size_t)mA * lda + (k0 + 32 + kA)], 0, 3);
        v16h a, b;
#pragma unroll
        for (int e = 0; e < 8; ++e) {
            a[e]     = (_Float16)Ap[(size_t)mA * lda + (k0 + kA + e)];
            a[e + 8] = (_Float16)Ap[(size_t)mA * lda + (k0 + 16 + kA + e)];
        }
#pragma unroll
        for (int e = 0; e < 16; ++e) {
            const int kk = k0 + kB + (e < 8 ? e : e + 8);
            b[e] = TRANSB ? (_Float16)Bp[(size_t)nB * ldb + kk]
                          : (_Float16)Bp[(size_t)kk * ldb + nB];
        }
        acc = __builtin_amdgcn_wmma_f32_16x16x32_f16(
            false, a, false, b, (short)0, acc, false, false);
    }

#pragma unroll
    for (int r = 0; r < 8; ++r) {
        const int m = m0 + r + 8 * (lane >> 4);
        const int n = n0 + (lane & 15);
        float val = alpha * acc[r];
        if constexpr (EPI == 1)
            val += Res[bb * sRb + hh * sRh + (size_t)m * ldc + n];
        if constexpr (EPI == 2) val = gelu_tanh(val);
        Cp[(size_t)m * ldc + n] = val;
    }
}

// ---------------------------------------------------------------------------
// One wave32 per 256-wide row: LayerNorm, softmax, squared-norm, VQ argmin.
// ---------------------------------------------------------------------------
__global__ __launch_bounds__(256)
void layernorm_k(const float* __restrict__ X, float* __restrict__ Y, int nrows)
{
    const int lane = threadIdx.x & 31;
    const int row  = blockIdx.x * 8 + (threadIdx.x >> 5);
    if (row >= nrows) return;
    const float* x = X + (size_t)row * D_;
    float v[8]; float s = 0.f;
#pragma unroll
    for (int j = 0; j < 8; ++j) { v[j] = x[lane + j * 32]; s += v[j]; }
#pragma unroll
    for (int m = 16; m; m >>= 1) s += __shfl_xor(s, m, 32);
    const float mu = s * (1.0f / 256.0f);
    float q = 0.f;
#pragma unroll
    for (int j = 0; j < 8; ++j) { const float d = v[j] - mu; q += d * d; }
#pragma unroll
    for (int m = 16; m; m >>= 1) q += __shfl_xor(q, m, 32);
    const float rstd = rsqrtf(q * (1.0f / 256.0f) + 1e-5f);
    float* y = Y + (size_t)row * D_;
#pragma unroll
    for (int j = 0; j < 8; ++j) y[lane + j * 32] = (v[j] - mu) * rstd;
}

__global__ __launch_bounds__(256)
void softmax_k(float* __restrict__ X, int nrows)
{
    const int lane = threadIdx.x & 31;
    const int row  = blockIdx.x * 8 + (threadIdx.x >> 5);
    if (row >= nrows) return;
    float* x = X + (size_t)row * S_;
    float v[8]; float mx = -3.4e38f;
#pragma unroll
    for (int j = 0; j < 8; ++j) { v[j] = x[lane + j * 32]; mx = fmaxf(mx, v[j]); }
#pragma unroll
    for (int m = 16; m; m >>= 1) mx = fmaxf(mx, __shfl_xor(mx, m, 32));
    float s = 0.f;
#pragma unroll
    for (int j = 0; j < 8; ++j) { v[j] = __expf(v[j] - mx); s += v[j]; }
#pragma unroll
    for (int m = 16; m; m >>= 1) s += __shfl_xor(s, m, 32);
    const float inv = 1.0f / s;
#pragma unroll
    for (int j = 0; j < 8; ++j) x[lane + j * 32] = v[j] * inv;
}

__global__ __launch_bounds__(256)
void rownorm_k(const float* __restrict__ X, float* __restrict__ nrm, int nrows)
{
    const int lane = threadIdx.x & 31;
    const int row  = blockIdx.x * 8 + (threadIdx.x >> 5);
    if (row >= nrows) return;
    const float* x = X + (size_t)row * D_;
    float s = 0.f;
#pragma unroll
    for (int j = 0; j < 8; ++j) { const float t = x[lane + j * 32]; s += t * t; }
#pragma unroll
    for (int m = 16; m; m >>= 1) s += __shfl_xor(s, m, 32);
    if (lane == 0) nrm[row] = s;
}

// min over a 2048-wide chunk of (||c||^2 - 2 z.c); ties -> lowest index (argmin).
__global__ __launch_bounds__(256)
void vq_argmin_k(const float* __restrict__ dot, const float* __restrict__ cnorm,
                 float* __restrict__ bestval, int* __restrict__ bestidx,
                 int c0, int chunk, int firstPass)
{
    const int lane = threadIdx.x & 31;
    const int row  = blockIdx.x * 8 + (threadIdx.x >> 5);
    if (row >= MROWS) return;
    float bv = 3.4e38f; int bi = 0x7fffffff;
    for (int n = lane; n < chunk; n += 32) {
        const float d = cnorm[c0 + n] - 2.0f * dot[(size_t)row * chunk + n];
        if (d < bv) { bv = d; bi = c0 + n; }
    }
#pragma unroll
    for (int m = 16; m; m >>= 1) {
        const float ov = __shfl_xor(bv, m, 32);
        const int   oi = __shfl_xor(bi, m, 32);
        if (ov < bv || (ov == bv && oi < bi)) { bv = ov; bi = oi; }
    }
    if (lane == 0) {
        if (firstPass || bv < bestval[row]) { bestval[row] = bv; bestidx[row] = bi; }
    }
}

__global__ __launch_bounds__(256)
void vq_gather_k(const int* __restrict__ bestidx, const float* __restrict__ codebook,
                 float* __restrict__ zq_out, int* __restrict__ c_out,
                 float* __restrict__ dec_in)
{
    const int row = blockIdx.x, t = threadIdx.x;
    const int idx = bestidx[row];
    const float v = codebook[(size_t)idx * D_ + t];
    zq_out[(size_t)row * D_ + t] = v;
    dec_in[(size_t)row * D_ + t] = v;
    if (t == 0) c_out[row] = idx;
}

__global__ __launch_bounds__(256)
void transpose_pe_k(const float* __restrict__ X, const float* __restrict__ pe,
                    float* __restrict__ Y)
{   // X:[B,D,S] -> Y:[B,S,D] + pe[S,D]
    const int i = blockIdx.x * blockDim.x + threadIdx.x;
    if (i >= B_ * S_ * D_) return;
    const int c = i % D_, s = (i / D_) % S_, b = i / (D_ * S_);
    Y[i] = X[(size_t)b * D_ * S_ + (size_t)c * S_ + s] + pe[s * D_ + c];
}

__global__ __launch_bounds__(256)
void transpose_out_k(const float* __restrict__ X, float* __restrict__ Y)
{   // X:[B,S,D] -> Y:[B,D,S]
    const int i = blockIdx.x * blockDim.x + threadIdx.x;
    if (i >= B_ * S_ * D_) return;
    const int s = i % S_, c = (i / S_) % D_, b = i / (D_ * S_);
    Y[i] = X[(size_t)b * S_ * D_ + (size_t)s * D_ + c];
}

// ---------------------------------------------------------------------------
// Host-side helpers
// ---------------------------------------------------------------------------
template<int EPI, bool TB>
static inline void gemmL(hipStream_t st, const float* A, const float* Bm,
                         const float* Res, float* C,
                         int M, int N, int Kd, int lda, int ldb, int ldc,
                         float alpha = 1.0f)
{
    dim3 g(N / 64, M / 128, 1), blk(256);
    gemm_wmma_lds<EPI, TB><<<g, blk, 0, st>>>(A, Bm, Res, C, M, N, Kd,
                                              lda, ldb, ldc, alpha);
}

template<int EPI, bool TB>
static inline void gemmT(hipStream_t st, const float* A, const float* Bm,
                         const float* Res, float* C,
                         int M, int N, int Kd, int lda, int ldb, int ldc,
                         int batches, int nh,
                         ll sAb, ll sAh, ll sBb, ll sBh,
                         ll sCb, ll sCh, ll sRb, ll sRh,
                         float alpha)
{
    dim3 g(N / 16, (M + 127) / 128, batches), blk(256);
    gemm_wmma<EPI, TB><<<g, blk, 0, st>>>(A, Bm, Res, C, M, N, Kd, lda, ldb, ldc,
                                          nh, sAb, sAh, sBb, sBh, sCb, sCh,
                                          sRb, sRh, alpha);
}

static void run_stack(hipStream_t st, float* x, const float* qkvo,
                      const float* ff1w, const float* ff2w,
                      float* h, float* q, float* k, float* v, float* o,
                      float* ff, float* sc)
{
    const float iscale = 0.17677669529f;   // 1/sqrt(hd=32)
    for (int l = 0; l < DEPTH_; ++l) {
        const float* wq = qkvo + ((size_t)l * 4 + 0) * D_ * D_;
        const float* wk = qkvo + ((size_t)l * 4 + 1) * D_ * D_;
        const float* wv = qkvo + ((size_t)l * 4 + 2) * D_ * D_;
        const float* wo = qkvo + ((size_t)l * 4 + 3) * D_ * D_;
        const float* f1 = ff1w + (size_t)l * D_ * DFF_;
        const float* f2 = ff2w + (size_t)l * DFF_ * D_;

        layernorm_k<<<MROWS / 8, 256, 0, st>>>(x, h, MROWS);
        gemmL<0, false>(st, h, wq, nullptr, q, MROWS, D_, D_, D_, D_, D_);
        gemmL<0, false>(st, h, wk, nullptr, k, MROWS, D_, D_, D_, D_, D_);
        gemmL<0, false>(st, h, wv, nullptr, v, MROWS, D_, D_, D_, D_, D_);
        // scores[b,h,q,k] = q . k^T / sqrt(hd)   (batched over B*H)
        gemmT<0, true >(st, q, k, nullptr, sc, S_, S_, HD_, D_, D_, S_,
                        B_ * H_, H_,
                        (ll)S_ * D_, (ll)HD_, (ll)S_ * D_, (ll)HD_,
                        (ll)H_ * S_ * S_, (ll)S_ * S_, 0, 0, iscale);
        softmax_k<<<(B_ * H_ * S_) / 8, 256, 0, st>>>(sc, B_ * H_ * S_);
        // o[b,s,h*hd] = attn . v
        gemmT<0, false>(st, sc, v, nullptr, o, S_, HD_, S_, S_, D_, D_,
                        B_ * H_, H_,
                        (ll)H_ * S_ * S_, (ll)S_ * S_, (ll)S_ * D_, (ll)HD_,
                        (ll)S_ * D_, (ll)HD_, 0, 0, 1.0f);
        // x = x + o @ wo
        gemmL<1, false>(st, o, wo, x, x, MROWS, D_, D_, D_, D_, D_);
        layernorm_k<<<MROWS / 8, 256, 0, st>>>(x, h, MROWS);
        // ff = gelu(h @ ff1)
        gemmL<2, false>(st, h, f1, nullptr, ff, MROWS, DFF_, D_, D_, DFF_, DFF_);
        // x = x + ff @ ff2
        gemmL<1, false>(st, ff, f2, x, x, MROWS, D_, DFF_, DFF_, D_, D_);
    }
}

extern "C" void kernel_launch(void* const* d_in, const int* in_sizes, int n_in,
                              void* d_out, int out_size, void* d_ws, size_t ws_size,
                              hipStream_t stream)
{
    (void)in_sizes; (void)n_in; (void)out_size; (void)ws_size;
    const float* x_in     = (const float*)d_in[0];
    const float* pe       = (const float*)d_in[1];
    const float* codebook = (const float*)d_in[2];
    const float* enc_qkvo = (const float*)d_in[3];
    const float* enc_ff1  = (const float*)d_in[4];
    const float* enc_ff2  = (const float*)d_in[5];
    const float* enc_proj = (const float*)d_in[6];
    const float* dec_qkvo = (const float*)d_in[7];
    const float* dec_ff1  = (const float*)d_in[8];
    const float* dec_ff2  = (const float*)d_in[9];
    const float* dec_proj = (const float*)d_in[10];

    float* out     = (float*)d_out;
    float* z_out   = out;                                   // [B,S,DC]
    float* zq_out  = out + (size_t)MROWS * D_;              // [B,S,DC]
    int*   c_out   = (int*)(out + 2 * (size_t)MROWS * D_);  // [B,S]
    float* rec_out = out + 2 * (size_t)MROWS * D_ + MROWS;  // [B,DOUT,S]

    // workspace carve-out (~76 MB of f32)
    float* ws = (float*)d_ws;
    size_t off = 0;
    auto carve = [&](size_t n) { float* p = ws + off; off += n; return p; };
    float* act   = carve((size_t)MROWS * D_);
    float* hbuf  = carve((size_t)MROWS * D_);
    float* qbuf  = carve((size_t)MROWS * D_);
    float* kbuf  = carve((size_t)MROWS * D_);
    float* vbuf  = carve((size_t)MROWS * D_);
    float* obuf  = carve((size_t)MROWS * D_);
    float* ffbuf = carve((size_t)MROWS * DFF_);
    float* scbuf = carve((size_t)B_ * H_ * S_ * S_);        // reused as VQ dot chunks
    float* cnorm = carve(K_);
    float* bestv = carve(MROWS);
    int*   besti = (int*)carve(MROWS);

    const int NELEM = B_ * S_ * D_;

    // z1 = transpose(x) + pe
    transpose_pe_k<<<(NELEM + 255) / 256, 256, 0, stream>>>(x_in, pe, act);
    // encoder
    run_stack(stream, act, enc_qkvo, enc_ff1, enc_ff2,
              hbuf, qbuf, kbuf, vbuf, obuf, ffbuf, scbuf);
    // z = enc_out @ enc_proj  (written straight to output, also VQ input)
    gemmL<0, false>(stream, act, enc_proj, nullptr, z_out, MROWS, D_, D_, D_, D_, D_);

    // VQ: argmin_k ||z - c_k||^2  ==  argmin_k (||c_k||^2 - 2 z.c_k)
    rownorm_k<<<K_ / 8, 256, 0, stream>>>(codebook, cnorm, K_);
    const int CH = 2048;
    for (int c0 = 0; c0 < K_; c0 += CH) {
        gemmL<0, true>(stream, z_out, codebook + (size_t)c0 * D_, nullptr, scbuf,
                       MROWS, CH, D_, D_, D_, CH);
        vq_argmin_k<<<MROWS / 8, 256, 0, stream>>>(scbuf, cnorm, bestv, besti,
                                                   c0, CH, c0 == 0 ? 1 : 0);
    }
    vq_gather_k<<<MROWS, 256, 0, stream>>>(besti, codebook, zq_out, c_out, act);

    // decoder
    run_stack(stream, act, dec_qkvo, dec_ff1, dec_ff2,
              hbuf, qbuf, kbuf, vbuf, obuf, ffbuf, scbuf);
    // rec = dec_out @ dec_proj, then [B,S,DOUT] -> [B,DOUT,S]
    gemmL<0, false>(stream, act, dec_proj, nullptr, ffbuf, MROWS, D_, D_, D_, D_, D_);
    transpose_out_k<<<(NELEM + 255) / 256, 256, 0, stream>>>(ffbuf, rec_out);
}
